// MoEFusionLayer_55387898249742
// MI455X (gfx1250) — matrix operand order repaired and checked by reading, not compiled, not run on hardware
//
#include <hip/hip_runtime.h>
#include <math.h>

#define B_ 8
#define L_ 1024
#define D_ 768
#define E_ 4
#define H_ 1536
#define PC_ 16   // pooling chunks over L

typedef __attribute__((ext_vector_type(16))) _Float16 v16h;
typedef __attribute__((ext_vector_type(8)))  _Float16 v8h;
typedef __attribute__((ext_vector_type(8)))  float    v8f;

// ---------------- prep kernels ----------------

// partial sums over L chunks: part[b][c][d] = sum_{l in chunk c} x[b][l][d]
__global__ void pool_part_kernel(const float* __restrict__ x, float* __restrict__ part) {
    int t = blockIdx.x * blockDim.x + threadIdx.x;
    if (t >= B_ * PC_ * D_) return;
    int d = t % D_;
    int r = t / D_;
    int c = r % PC_;
    int b = r / PC_;
    const float* p = x + ((size_t)b * L_ + (size_t)c * (L_ / PC_)) * D_ + d;
    float s = 0.f;
    for (int l = 0; l < L_ / PC_; ++l) s += p[(size_t)l * D_];
    part[t] = s;
}

__global__ void router_kernel(const float* __restrict__ part,
                              const float* __restrict__ rw,
                              const float* __restrict__ rb,
                              float* __restrict__ gate) {
    __shared__ float logits[B_][E_];
    int t = threadIdx.x;
    if (t < B_ * E_) {
        int b = t / E_, e = t % E_;
        const float* pp = part + (size_t)b * PC_ * D_;
        const float* wv = rw + e * D_;
        float s = rb[e];
        for (int d = 0; d < D_; ++d) {
            float pd = 0.f;
            for (int c = 0; c < PC_; ++c) pd += pp[(size_t)c * D_ + d];
            s += (pd * (1.0f / L_)) * wv[d];
        }
        logits[b][e] = s;
    }
    __syncthreads();
    if (t < B_) {
        int i0 = 0; float v0 = logits[t][0];
        for (int e = 1; e < E_; ++e) if (logits[t][e] > v0) { v0 = logits[t][e]; i0 = e; }
        int i1 = -1; float v1 = -INFINITY;
        for (int e = 0; e < E_; ++e) { if (e == i0) continue; if (logits[t][e] > v1) { v1 = logits[t][e]; i1 = e; } }
        float e0 = 1.0f, e1 = expf(v1 - v0);
        float inv = 1.f / (e0 + e1);
        float g[E_] = {0.f, 0.f, 0.f, 0.f};
        g[i0] = e0 * inv; g[i1] = e1 * inv;
        for (int e = 0; e < E_; ++e) gate[t * E_ + e] = g[e];
    }
}

__global__ void cvt_f16_kernel(const float* __restrict__ src, _Float16* __restrict__ dst, long n) {
    long i = (long)blockIdx.x * blockDim.x + threadIdx.x;
    if (i < n) dst[i] = (_Float16)src[i];
}

// Repack src [E_, K, N] fp32 into per-lane WMMA B-fragment images:
// dst[e][nt][kt][lane][16 f16]; element (v,hh): k = kt*32 + v*2 + hh + (v>=4)*8 + (lane>>4)*8, n = nt*16 + (lane&15)
__global__ void pack_b_kernel(const float* __restrict__ src, _Float16* __restrict__ dst,
                              int K, int N) {
    int nkt = K >> 5, nnt = N >> 4;
    long tid = (long)blockIdx.x * blockDim.x + threadIdx.x;
    long total = (long)E_ * nnt * nkt * 32;
    if (tid >= total) return;
    int lane = (int)(tid & 31);
    long r = tid >> 5;
    int kt = (int)(r % nkt); r /= nkt;
    int nt = (int)(r % nnt);
    int e  = (int)(r / nnt);
    int n  = nt * 16 + (lane & 15);
    int lh = lane >> 4;
    _Float16* out = dst + tid * 16;
    const float* s = src + (long)e * K * N + n;
#pragma unroll
    for (int v = 0; v < 8; ++v)
#pragma unroll
        for (int hh = 0; hh < 2; ++hh) {
            int k = kt * 32 + v * 2 + hh + (v >= 4 ? 8 : 0) + lh * 8;
            out[v * 2 + hh] = (_Float16)s[(long)k * N];
        }
}

// ---------------- main fused MoE kernel ----------------

__device__ __forceinline__ float act_gelu(float v) {
    return 0.5f * v * (1.0f + erff(v * 0.70710678118654752440f));
}
__device__ __forceinline__ float act_silu(float v) {
    return v / (1.0f + expf(-v));
}
__device__ __forceinline__ float act_mish(float v) {
    float sp = (v > 20.0f) ? v : log1pf(expf(v));
    return v * tanhf(sp);
}

// byte strides in packed weight buffers (per v16h fragment = 1024B across 32 lanes)
#define W1_NT_B   (24 * 1024)          // one H-tile step: nkt(24) fragments
#define W1_HK_B   (48 * W1_NT_B)       // 48 H-tiles per chunk
#define W1_E_B    (96 * W1_NT_B)       // 96 H-tiles per expert
#define W2_NT_B   (48 * 1024)          // one D-tile step: nkt(48) fragments
#define W2_HK_B   (24 * 1024)          // 24 K-tiles per H-chunk
#define W2_E_B    (48 * W2_NT_B)

// block = 256 threads (8 waves), handles 16 rows (one sample b, rows m0..m0+15)
__global__ __launch_bounds__(256)
void moe_main_kernel(const float* __restrict__ x,
                     const _Float16* __restrict__ xh,
                     const _Float16* __restrict__ w1p,
                     const _Float16* __restrict__ w2p,
                     const float* __restrict__ b1,
                     const float* __restrict__ b2,
                     const float* __restrict__ gate,
                     const float* __restrict__ lng,
                     const float* __restrict__ lnb,
                     float* __restrict__ out) {
    // LDS: sx = x tile [16][776] f16 (24832B), sh = h chunk [16][776] f16 (24832B)
    // after GEMMs the whole buffer is reused as y [16][772] f32 (49408B)
    __shared__ __align__(16) unsigned char smem[24832 * 2];
    _Float16 (*sx)[776] = (_Float16(*)[776])smem;
    _Float16 (*sh)[776] = (_Float16(*)[776])(smem + 24832);
    float* sy = (float*)smem;

    const int b    = blockIdx.x >> 6;
    const int m0   = (blockIdx.x & 63) << 4;
    const int tid  = threadIdx.x;
    const int w    = tid >> 5;
    const int lane = tid & 31;
    const int ln16 = lane & 15;
    const int lh   = lane >> 4;

    // phase 0: cooperative copy of x tile (f16) into LDS (tile is contiguous in xh)
    {
        const uint4* src = (const uint4*)(xh + ((size_t)b * L_ + m0) * D_);
        for (int i = tid; i < 16 * (D_ / 8); i += 256) {
            int r = i / (D_ / 8), c = i % (D_ / 8);
            *(uint4*)&sx[r][c * 8] = src[(size_t)r * (D_ / 8) + c];
        }
    }
    __syncthreads();

    v8f facc[6];
#pragma unroll
    for (int j = 0; j < 6; ++j)
#pragma unroll
        for (int v = 0; v < 8; ++v) facc[j][v] = 0.0f;

    // per-lane A-fragment bases (immediate offsets added in the loops)
    const char* axb = (const char*)&sx[ln16][0] + lh * 16;
    const char* ahb = (const char*)&sh[ln16][0] + lh * 16;

    // per-(wave,lane) packed-weight bases; expert/chunk offsets added below
    const char* w1base0 = (const char*)w1p + (size_t)w * 6 * W1_NT_B + (size_t)lane * 32;
    const char* w2base0 = (const char*)w2p + (size_t)w * 6 * W2_NT_B + (size_t)lane * 32;

    for (int e = 0; e < E_; ++e) {
        const float ge = gate[b * E_ + e];
        if (ge == 0.0f) continue;   // uniform across block

        const char* w1be = w1base0 + (size_t)e * W1_E_B;
        const char* w2be = w2base0 + (size_t)e * W2_E_B;

        v8f oacc[6];
#pragma unroll
        for (int j = 0; j < 6; ++j)
#pragma unroll
            for (int v = 0; v < 8; ++v) oacc[j][v] = 0.0f;

        for (int hk = 0; hk < 2; ++hk) {   // H processed in 2 chunks of 768
            __syncthreads();               // protect sh before rewrite
            const char* w1b = w1be + (size_t)hk * W1_HK_B;
            const char* w2b = w2be + (size_t)hk * W2_HK_B;

            // ---- GEMM1 chunk: h[:, hk*768 .. +768] = act(x @ w1 + b1) ----
            {
                v8f hacc[6];
#pragma unroll
                for (int j = 0; j < 6; ++j)
#pragma unroll
                    for (int v = 0; v < 8; ++v) hacc[j][v] = 0.0f;

#pragma unroll 4
                for (int kt = 0; kt < D_ / 32; ++kt) {
                    const v8h a0 = *(const v8h*)(axb + kt * 64);
                    const v8h a1 = *(const v8h*)(axb + kt * 64 + 32);
                    v16h A = __builtin_shufflevector(a0, a1,
                        0,1,2,3,4,5,6,7,8,9,10,11,12,13,14,15);
#pragma unroll
                    for (int j = 0; j < 6; ++j) {
                        v16h Bm = *(const v16h*)(w1b + j * W1_NT_B + kt * 1024);
                        hacc[j] = __builtin_amdgcn_wmma_f32_16x16x32_f16(
                            false, A, false, Bm, (short)0, hacc[j], false, false);
                    }
                }
                // bias + activation + store f16 chunk to LDS
#pragma unroll
                for (int j = 0; j < 6; ++j) {
                    int ntl = w * 6 + j;                 // local chunk tile
                    int ntg = hk * 48 + ntl;             // global H tile
                    float bias = b1[e * H_ + ntg * 16 + ln16];
#pragma unroll
                    for (int v = 0; v < 8; ++v) {
                        float f = hacc[j][v] + bias;
                        if (e == 1)      f = act_silu(f);
                        else if (e == 2) f = act_mish(f);
                        else             f = act_gelu(f);
                        sh[v + 8 * lh][ntl * 16 + ln16] = (_Float16)f;
                    }
                }
            }
            __syncthreads();

            // ---- GEMM2 partial: oacc += h_chunk @ w2[hk*768.. , :] ----
#pragma unroll 4
            for (int kt2 = 0; kt2 < 24; ++kt2) {
                const v8h a0 = *(const v8h*)(ahb + kt2 * 64);
                const v8h a1 = *(const v8h*)(ahb + kt2 * 64 + 32);
                v16h A = __builtin_shufflevector(a0, a1,
                    0,1,2,3,4,5,6,7,8,9,10,11,12,13,14,15);
#pragma unroll
                for (int j = 0; j < 6; ++j) {
                    v16h Bm = *(const v16h*)(w2b + j * W2_NT_B + kt2 * 1024);
                    oacc[j] = __builtin_amdgcn_wmma_f32_16x16x32_f16(
                        false, A, false, Bm, (short)0, oacc[j], false, false);
                }
            }
        }

        // gated accumulate (+ b2)
#pragma unroll
        for (int j = 0; j < 6; ++j) {
            int nt = w * 6 + j;
            float b2v = b2[e * D_ + nt * 16 + ln16];
#pragma unroll
            for (int v = 0; v < 8; ++v)
                facc[j][v] += ge * (oacc[j][v] + b2v);
        }
    }

    __syncthreads();   // done with sx/sh; reuse as sy

    // residual (fp32 x from global) and stage y into LDS
#pragma unroll
    for (int j = 0; j < 6; ++j) {
        int nt = w * 6 + j;
#pragma unroll
        for (int v = 0; v < 8; ++v) {
            int m = v + 8 * lh;
            int c = nt * 16 + ln16;
            float yv = facc[j][v] + x[((size_t)b * L_ + m0 + m) * D_ + c];
            sy[(size_t)m * 772 + c] = yv;
        }
    }
    __syncthreads();

    // LayerNorm: 2 rows per wave, wave32 shuffle reduction
    for (int rr = 0; rr < 2; ++rr) {
        int r = w * 2 + rr;
        const float* row = &sy[(size_t)r * 772];
        float s = 0.f, q = 0.f;
        for (int c = lane; c < D_; c += 32) { float v = row[c]; s += v; q += v * v; }
#pragma unroll
        for (int o = 16; o > 0; o >>= 1) { s += __shfl_xor(s, o); q += __shfl_xor(q, o); }
        float mu  = s * (1.0f / D_);
        float var = q * (1.0f / D_) - mu * mu;
        float rstd = rsqrtf(var + 1e-5f);
        float* op = out + ((size_t)b * L_ + m0 + r) * D_;
        for (int c = lane; c < D_; c += 32)
            op[c] = (row[c] - mu) * rstd * lng[c] + lnb[c];
    }
}

// ---------------- launch ----------------

extern "C" void kernel_launch(void* const* d_in, const int* in_sizes, int n_in,
                              void* d_out, int out_size, void* d_ws, size_t ws_size,
                              hipStream_t stream) {
    const float* x   = (const float*)d_in[0];
    const float* rw  = (const float*)d_in[1];
    const float* rb  = (const float*)d_in[2];
    const float* w1  = (const float*)d_in[3];
    const float* b1  = (const float*)d_in[4];
    const float* w2  = (const float*)d_in[5];
    const float* b2  = (const float*)d_in[6];
    const float* lng = (const float*)d_in[7];
    const float* lnb = (const float*)d_in[8];
    float* out = (float*)d_out;

    unsigned char* ws = (unsigned char*)d_ws;
    size_t off = 0;
    auto take = [&](size_t bytes) { unsigned char* p = ws + off; off = (off + bytes + 255) & ~(size_t)255; return p; };
    float*    part = (float*)   take((size_t)B_ * PC_ * D_ * 4);
    float*    gate = (float*)   take((size_t)B_ * E_ * 4);
    _Float16* xh   = (_Float16*)take((size_t)B_ * L_ * D_ * 2);
    _Float16* w1p  = (_Float16*)take((size_t)E_ * D_ * H_ * 2);
    _Float16* w2p  = (_Float16*)take((size_t)E_ * H_ * D_ * 2);
    (void)ws_size; (void)in_sizes; (void)n_in; (void)out_size;

    long nx = (long)B_ * L_ * D_;
    cvt_f16_kernel<<<(int)((nx + 255) / 256), 256, 0, stream>>>(x, xh, nx);

    long nw1 = (long)E_ * (H_ / 16) * (D_ / 32) * 32;
    pack_b_kernel<<<(int)((nw1 + 255) / 256), 256, 0, stream>>>(w1, w1p, D_, H_);
    long nw2 = (long)E_ * (D_ / 16) * (H_ / 32) * 32;
    pack_b_kernel<<<(int)((nw2 + 255) / 256), 256, 0, stream>>>(w2, w2p, H_, D_);

    pool_part_kernel<<<(B_ * PC_ * D_ + 255) / 256, 256, 0, stream>>>(x, part);
    router_kernel<<<1, 256, 0, stream>>>(part, rw, rb, gate);

    moe_main_kernel<<<B_ * (L_ / 16), 256, 0, stream>>>(
        x, xh, w1p, w2p, b1, b2, gate, lng, lnb, out);
}
